// ScigptMoeDecoderLayerPP_59167469470458
// MI455X (gfx1250) — matrix operand order repaired and checked by reading, not compiled, not run on hardware
//
#include <hip/hip_runtime.h>
#include <cstdint>
#include <cstddef>

typedef __bf16 bf16;
typedef __attribute__((ext_vector_type(16))) __bf16 bf16x16;
typedef __attribute__((ext_vector_type(8)))  __bf16 bf16x8;
typedef __attribute__((ext_vector_type(8)))  float  f32x8;

constexpr int CB = 2, CS = 2048, CH = 1024, CNH = 16, CNKV = 4, CHD = 64;
constexpr int CE = 8, CF = 2048, CT = CB * CS;
constexpr int RMAX = 8448;           // 8192 gathered rows + per-expert pad to 32
constexpr int MAXBT = RMAX / 32;     // 264 row block-tiles

#if defined(__HIP_DEVICE_COMPILE__) && __has_builtin(__builtin_amdgcn_tensor_load_to_lds)
#define HAVE_TDM 1
#else
#define HAVE_TDM 0
#endif

// ---------------------------------------------------------------- helpers ----
__device__ __forceinline__ bf16x16 cat8(bf16x8 lo, bf16x8 hi) {
  return __builtin_shufflevector(lo, hi, 0,1,2,3,4,5,6,7,8,9,10,11,12,13,14,15);
}
__device__ __forceinline__ f32x8 wmma_bf16(bf16x16 a, bf16x16 b, f32x8 c) {
  return __builtin_amdgcn_wmma_f32_16x16x32_bf16(false, a, false, b, (short)0, c,
                                                 false, false);
}
__device__ __forceinline__ f32x8 zero8() {
  f32x8 z;
#pragma unroll
  for (int i = 0; i < 8; ++i) z[i] = 0.f;
  return z;
}

#if HAVE_TDM
typedef __attribute__((ext_vector_type(4))) unsigned u32x4;
typedef __attribute__((ext_vector_type(8))) int      i32x8;
typedef __attribute__((ext_vector_type(4))) int      i32x4;

// 2D TDM tile load: tile_d0 elements (contiguous, 2B each) x tile_d1 lines.
// pad_enable: +4 DWORDs after every 16 DWORDs -> LDS line stride 80B
// (20-bank stride => conflict-free ds_load_b128 across 16 lanes, 16B aligned).
__device__ __forceinline__ void tdm_load_tile_2d(
    unsigned lds_addr, const void* gptr, unsigned tensor_d0, unsigned tensor_d1,
    unsigned tile_d0, unsigned tile_d1, unsigned long long stride0) {
  unsigned long long ga = (unsigned long long)(uintptr_t)gptr;
  u32x4 g0;
  g0[0] = 1u;                                   // count=1 (valid), user mode
  g0[1] = lds_addr;                             // LDS byte address
  g0[2] = (unsigned)ga;                         // global_addr[31:0]
  g0[3] = (unsigned)((ga >> 32) & 0x01ffffffu) | (2u << 30);  // addr[56:32] | type=2
  i32x8 g1;
  g1[0] = (int)((1u << 16)      // data_size = 1 -> 2 bytes
              | (1u << 20)      // pad_enable
              | (3u << 22)      // pad_interval: every 16 DWORDs (64B)
              | (3u << 25));    // pad_amount: 4 DWORDs (16B)
  g1[1] = (int)((tensor_d0 & 0xffffu) << 16);                        // td0[15:0]
  g1[2] = (int)((tensor_d0 >> 16) | ((tensor_d1 & 0xffffu) << 16));  // td0 hi | td1 lo
  g1[3] = (int)((tensor_d1 >> 16) | ((tile_d0 & 0xffffu) << 16));    // td1 hi | tile0
  g1[4] = (int)(tile_d1 & 0xffffu);                                  // tile1 | tile2=0
  g1[5] = (int)(stride0 & 0xffffffffu);                              // dim0 stride lo
  g1[6] = (int)((stride0 >> 32) & 0xffffu);                          // stride hi | s1 lo
  g1[7] = 0;
  i32x4 gz  = {0, 0, 0, 0};
  i32x8 gz8 = {0, 0, 0, 0, 0, 0, 0, 0};
  // 6-arg form (this toolchain): (g0, g1, g2, g3, g4, cpol)
  __builtin_amdgcn_tensor_load_to_lds(g0, g1, gz, gz, gz8, 0);
}
#endif

// -------------------------------------------------- weight convert+transpose --
// in: fp32 [batch][K][N] row-major -> out: bf16 [batch][N][K] (K contiguous)
__global__ __launch_bounds__(256) void transpose_convert_kernel(
    const float* __restrict__ in, bf16* __restrict__ out, int batch, int K, int N) {
  size_t idx = (size_t)blockIdx.x * blockDim.x + threadIdx.x;
  size_t total = (size_t)batch * K * N;
  if (idx >= total) return;
  size_t b = idx / ((size_t)K * N);
  size_t r = idx % ((size_t)K * N);
  size_t k = r / N, n = r % N;
  out[b * (size_t)K * N + n * (size_t)K + k] = (bf16)in[idx];
}

__global__ __launch_bounds__(256) void zero_u32_kernel(unsigned* __restrict__ p, size_t n) {
  size_t i = (size_t)blockIdx.x * blockDim.x + threadIdx.x;
  if (i < n) p[i] = 0u;
}

// ---------------------------------------------------------------- rmsnorm ----
__global__ __launch_bounds__(256) void rmsnorm_kernel(
    const float* __restrict__ x, const float* __restrict__ w, bf16* __restrict__ out) {
  const int t = blockIdx.x;
  __shared__ float red[256];
  const float* xr = x + (size_t)t * CH;
  float s = 0.f;
  for (int j = threadIdx.x; j < CH; j += 256) { float v = xr[j]; s += v * v; }
  red[threadIdx.x] = s;
  __syncthreads();
  for (int o = 128; o > 0; o >>= 1) {
    if (threadIdx.x < o) red[threadIdx.x] += red[threadIdx.x + o];
    __syncthreads();
  }
  float inv = rsqrtf(red[0] / (float)CH + 1e-5f);
  for (int j = threadIdx.x; j < CH; j += 256)
    out[(size_t)t * CH + j] = (bf16)(xr[j] * inv * w[j]);
}

// -------------------------------------------------------------- WMMA GEMM ----
// C[M,N] = A[M,K](bf16,row-major) * BT[N,K](bf16)   EPI: 0=f32 store, 1=+resid
template <int EPI>
__global__ __launch_bounds__(256) void gemm_bf16_kernel(
    const bf16* __restrict__ A, const bf16* __restrict__ BT,
    float* __restrict__ Cf, const float* __restrict__ resid, int M, int N, int K) {
  const int wave = threadIdx.x >> 5, lane = threadIdx.x & 31;
  const int half = lane >> 4, ln = lane & 15;
  const int row0 = blockIdx.y * 32 + (wave >> 2) * 16;
  const int col0 = blockIdx.x * 256 + (wave & 3) * 64;
  if (row0 >= M || col0 >= N) return;

  f32x8 acc[4];
#pragma unroll
  for (int j = 0; j < 4; ++j) acc[j] = zero8();

  const bf16* aRow = A + (size_t)(row0 + ln) * K;
  for (int k0 = 0; k0 < K; k0 += 32) {
    bf16x16 a = cat8(*(const bf16x8*)(aRow + k0 + half * 16),
                     *(const bf16x8*)(aRow + k0 + half * 16 + 8));
#pragma unroll
    for (int j = 0; j < 4; ++j) {
      const bf16* bp = BT + (size_t)(col0 + j * 16 + ln) * K + k0 + half * 16;
      __builtin_prefetch(bp + 32, 0, 3);
      bf16x16 bb = cat8(*(const bf16x8*)bp, *(const bf16x8*)(bp + 8));
      acc[j] = wmma_bf16(a, bb, acc[j]);
    }
  }
#pragma unroll
  for (int j = 0; j < 4; ++j)
#pragma unroll
    for (int r = 0; r < 8; ++r) {
      size_t o = (size_t)(row0 + r + 8 * half) * N + col0 + j * 16 + ln;
      if constexpr (EPI == 1) Cf[o] = acc[j][r] + resid[o];
      else Cf[o] = acc[j][r];
    }
}

// ------------------------------------------------------------------- RoPE ----
__global__ __launch_bounds__(256) void rope_kernel(
    const float* __restrict__ qf, const int* __restrict__ pos, bf16* __restrict__ outb,
    int nheads, float scale) {
  int idx = blockIdx.x * blockDim.x + threadIdx.x;
  int total = CT * nheads * (CHD / 2);
  if (idx >= total) return;
  int i = idx & 31;
  int hh = (idx >> 5) % nheads;
  int t = idx / (32 * nheads);
  float p = (float)pos[t];
  float ang = p * __expf(-(float)i * (13.815510558f / 32.f));  // theta=1e6
  float sn, cs;
  __sincosf(ang, &sn, &cs);
  size_t base = ((size_t)t * nheads + hh) * CHD;
  float x1 = qf[base + i], x2 = qf[base + 32 + i];
  outb[base + i]      = (bf16)((x1 * cs - x2 * sn) * scale);
  outb[base + 32 + i] = (bf16)((x2 * cs + x1 * sn) * scale);
}

// V: [B,S,NKV,HD] f32 -> [B,NKV,HD,S] bf16 (key-contiguous for PV B-fragments)
__global__ __launch_bounds__(256) void vtrans_kernel(const float* __restrict__ vf,
                                                     bf16* __restrict__ vbT) {
  int idx = blockIdx.x * blockDim.x + threadIdx.x;
  int total = CT * CNKV * CHD;
  if (idx >= total) return;
  int d = idx & 63, kv = (idx >> 6) & 3, t = idx >> 8;
  int b = t / CS, s = t % CS;
  vbT[(((size_t)b * CNKV + kv) * CHD + d) * CS + s] = (bf16)vf[idx];
}

// -------------------------------------------------------- flash attention ----
__global__ __launch_bounds__(32) void flash_attn_kernel(
    const bf16* __restrict__ qb, const bf16* __restrict__ kb,
    const bf16* __restrict__ vbT, bf16* __restrict__ attn) {
  const int lane = threadIdx.x;
  const int qt = blockIdx.x, h = blockIdx.y, b = blockIdx.z;
  const int kvh = h >> 2;
  const int half = lane >> 4, ln = lane & 15;
  __shared__ bf16 pTile[16][40];

  bf16x16 aq0, aq1;
  {
    const bf16* p = qb + (((size_t)b * CS + qt * 16 + ln) * CNH + h) * CHD;
    aq0 = cat8(*(const bf16x8*)(p + half * 16), *(const bf16x8*)(p + half * 16 + 8));
    aq1 = cat8(*(const bf16x8*)(p + 32 + half * 16), *(const bf16x8*)(p + 40 + half * 16));
  }

  float m[8], l[8];
  f32x8 accO[4];
#pragma unroll
  for (int r = 0; r < 8; ++r) { m[r] = -1e30f; l[r] = 0.f; }
#pragma unroll
  for (int j = 0; j < 4; ++j) accO[j] = zero8();

  const int qbase = qt * 16;
  for (int k32 = 0; k32 <= qbase; k32 += 32) {
    f32x8 sA = zero8(), sB = zero8();
    {
      const bf16* pk = kb + (((size_t)b * CS + k32 + ln) * CNKV + kvh) * CHD;
      bf16x16 b0 = cat8(*(const bf16x8*)(pk + half * 16),
                        *(const bf16x8*)(pk + half * 16 + 8));
      bf16x16 b1 = cat8(*(const bf16x8*)(pk + 32 + half * 16),
                        *(const bf16x8*)(pk + 32 + half * 16 + 8));
      sA = wmma_bf16(aq0, b0, sA);
      sA = wmma_bf16(aq1, b1, sA);
      const bf16* pk2 = pk + (size_t)16 * CNKV * CHD;
      b0 = cat8(*(const bf16x8*)(pk2 + half * 16),
                *(const bf16x8*)(pk2 + half * 16 + 8));
      b1 = cat8(*(const bf16x8*)(pk2 + 32 + half * 16),
                *(const bf16x8*)(pk2 + 32 + half * 16 + 8));
      sB = wmma_bf16(aq0, b0, sB);
      sB = wmma_bf16(aq1, b1, sB);
    }
    if (k32 + 31 > qbase) {  // causal mask only near the diagonal
#pragma unroll
      for (int r = 0; r < 8; ++r) {
        int qg = qbase + r + 8 * half;
        if (k32 + ln > qg)      sA[r] = -1e30f;
        if (k32 + 16 + ln > qg) sB[r] = -1e30f;
      }
    }
#pragma unroll
    for (int r = 0; r < 8; ++r) {
      float mx = fmaxf(sA[r], sB[r]);
      mx = fmaxf(mx, __shfl_xor(mx, 1));
      mx = fmaxf(mx, __shfl_xor(mx, 2));
      mx = fmaxf(mx, __shfl_xor(mx, 4));
      mx = fmaxf(mx, __shfl_xor(mx, 8));
      float mn = fmaxf(m[r], mx);
      float sc = __expf(m[r] - mn);
      m[r] = mn;
      float pA = __expf(sA[r] - mn);
      float pB = __expf(sB[r] - mn);
      float rs = pA + pB;
      rs += __shfl_xor(rs, 1);
      rs += __shfl_xor(rs, 2);
      rs += __shfl_xor(rs, 4);
      rs += __shfl_xor(rs, 8);
      l[r] = l[r] * sc + rs;
      pTile[r + 8 * half][ln]      = (bf16)pA;
      pTile[r + 8 * half][16 + ln] = (bf16)pB;
#pragma unroll
      for (int j = 0; j < 4; ++j) accO[j][r] = accO[j][r] * sc;
    }
    __syncthreads();
    bf16x16 ap;
#pragma unroll
    for (int i = 0; i < 8; ++i) {          // C-layout -> A-layout transpose via LDS
      ap[i]     = pTile[ln][half * 16 + i];
      ap[8 + i] = pTile[ln][half * 16 + 8 + i];
    }
    __syncthreads();
#pragma unroll
    for (int j = 0; j < 4; ++j) {
      const bf16* pv = vbT + (((size_t)b * CNKV + kvh) * CHD + j * 16 + ln) * CS +
                       k32 + half * 16;
      bf16x16 bv = cat8(*(const bf16x8*)pv, *(const bf16x8*)(pv + 8));
      accO[j] = wmma_bf16(ap, bv, accO[j]);
    }
  }
#pragma unroll
  for (int j = 0; j < 4; ++j)
#pragma unroll
    for (int r = 0; r < 8; ++r) {
      size_t o = (((size_t)b * CS + qbase + r + 8 * half) * CNH + h) * CHD + j * 16 + ln;
      attn[o] = (bf16)(accO[j][r] / l[r]);
    }
}

// ------------------------------------------------------------------ router ----
__global__ __launch_bounds__(256) void router_kernel(
    const bf16* __restrict__ x2, const float* __restrict__ rw,
    float* __restrict__ gate_out, int* __restrict__ topi, float* __restrict__ topw,
    int* __restrict__ counts) {
  const int t = blockIdx.x;
  const int wave = threadIdx.x >> 5, lane = threadIdx.x & 31;  // wave == expert
  __shared__ float lg[CE];
  const bf16* xr = x2 + (size_t)t * CH;
  float s = 0.f;
  for (int j = lane; j < CH; j += 32) s += (float)xr[j] * rw[(size_t)j * CE + wave];
  s += __shfl_xor(s, 1);
  s += __shfl_xor(s, 2);
  s += __shfl_xor(s, 4);
  s += __shfl_xor(s, 8);
  s += __shfl_xor(s, 16);
  if (lane == 0) { lg[wave] = s; gate_out[(size_t)t * CE + wave] = s; }
  __syncthreads();
  if (threadIdx.x == 0) {
    float mx = lg[0];
    for (int e = 1; e < CE; ++e) mx = fmaxf(mx, lg[e]);
    float p[CE], sum = 0.f;
    for (int e = 0; e < CE; ++e) { p[e] = __expf(lg[e] - mx); sum += p[e]; }
    for (int e = 0; e < CE; ++e) p[e] /= sum;
    int i1 = 0;
    for (int e = 1; e < CE; ++e) if (p[e] > p[i1]) i1 = e;
    int i2 = (i1 == 0) ? 1 : 0;
    for (int e = 0; e < CE; ++e) if (e != i1 && p[e] > p[i2]) i2 = e;
    float tot = p[i1] + p[i2];
    topi[t * 2] = i1;       topi[t * 2 + 1] = i2;
    topw[t * 2] = p[i1] / tot; topw[t * 2 + 1] = p[i2] / tot;
    atomicAdd(&counts[i1], 1);
    atomicAdd(&counts[i2], 1);
  }
}

__global__ void offsets_kernel(const int* __restrict__ counts, int* __restrict__ base,
                               int* __restrict__ tile_expert) {
  if (blockIdx.x == 0 && threadIdx.x == 0) {
    int pb = 0;
    for (int e = 0; e < CE; ++e) {
      base[e] = pb;
      int pc = (counts[e] + 31) & ~31;
      for (int bt = pb >> 5; bt < (pb + pc) >> 5; ++bt) tile_expert[bt] = e;
      pb += pc;
    }
    for (int bt = pb >> 5; bt < MAXBT; ++bt) tile_expert[bt] = -1;
  }
}

__global__ __launch_bounds__(256) void scatter_kernel(
    const int* __restrict__ topi, const int* __restrict__ base, int* __restrict__ fill,
    int* __restrict__ tok2row) {
  int idx = blockIdx.x * blockDim.x + threadIdx.x;
  if (idx >= CT * 2) return;
  int e = topi[idx];
  int ppos = base[e] + atomicAdd(&fill[e], 1);
  tok2row[idx] = ppos;
}

__global__ __launch_bounds__(256) void gather_kernel(
    const bf16* __restrict__ x2, const int* __restrict__ tok2row, bf16* __restrict__ Xg) {
  int idx = blockIdx.x;            // token-slot 0..2T-1
  int row = tok2row[idx];
  int t = idx >> 1;
  for (int j = threadIdx.x; j < CH; j += 256)
    Xg[(size_t)row * CH + j] = x2[(size_t)t * CH + j];
}

// ----------------------------------------------------- MoE expert GEMMs ------
// w1/w3 fused GEMM + SiLU*mul.  B tiles (256 cols x 32 K, 16KB each) staged to
// LDS by the Tensor Data Mover, double-buffered on TENSORcnt; padded 80B lines
// give a 20-bank stride (conflict-free ds_load_b128 across the 16 frag lanes).
__global__ __launch_bounds__(256) void moe_gemm13_kernel(
    const bf16* __restrict__ Xg, const bf16* __restrict__ w1T,
    const bf16* __restrict__ w3T, const int* __restrict__ tile_expert,
    bf16* __restrict__ Act) {
  const int e = tile_expert[blockIdx.y];
  if (e < 0) return;
  const int wave = threadIdx.x >> 5, lane = threadIdx.x & 31;
  const int half = lane >> 4, ln = lane & 15;
  const int row0 = blockIdx.y * 32 + (wave >> 2) * 16;
  const int colB = blockIdx.x * 256;
  const int col0loc = (wave & 3) * 64;

  const bf16* B1 = w1T + (size_t)e * CF * CH + (size_t)colB * CH;
  const bf16* B3 = w3T + (size_t)e * CF * CH + (size_t)colB * CH;

  f32x8 a1[4], a3[4];
#pragma unroll
  for (int j = 0; j < 4; ++j) { a1[j] = zero8(); a3[j] = zero8(); }
  const bf16* aRow = Xg + (size_t)(row0 + ln) * CH;

#if HAVE_TDM
  constexpr int LINEB = 80;              // 64B data + 16B TDM pad per line
  constexpr int TILEB = 256 * LINEB;     // 20KB per matrix tile
  __shared__ __attribute__((aligned(128))) char ldsB[2][2][TILEB];
  const int NIT = CH / 32;
  const unsigned lds0 = (unsigned)(uintptr_t)&ldsB[0][0][0];  // LDS addr = low 32b
  if (wave == 0) {  // prologue: fill both buffers (tiles 0 and 1)
    tdm_load_tile_2d(lds0 + 0u * TILEB, B1,      CH, CF, 32, 256, CH);
    tdm_load_tile_2d(lds0 + 1u * TILEB, B3,      CH, CF, 32, 256, CH);
    tdm_load_tile_2d(lds0 + 2u * TILEB, B1 + 32, CH, CF, 32, 256, CH);
    tdm_load_tile_2d(lds0 + 3u * TILEB, B3 + 32, CH, CF, 32, 256, CH);
  }
  for (int it = 0; it < NIT; ++it) {
    const int k0 = it * 32;
    if (wave == 0) {
      if (it < NIT - 1) __builtin_amdgcn_s_wait_tensorcnt(2);
      else              __builtin_amdgcn_s_wait_tensorcnt(0);
    }
    __syncthreads();                     // buffer (it&1) is ready for all waves
    bf16x16 a = cat8(*(const bf16x8*)(aRow + k0 + half * 16),
                     *(const bf16x8*)(aRow + k0 + half * 16 + 8));
    const char* b1buf = &ldsB[it & 1][0][0];
    const char* b3buf = &ldsB[it & 1][1][0];
#pragma unroll
    for (int j = 0; j < 4; ++j) {
      const int lc = col0loc + j * 16 + ln;
      const bf16* p1 = (const bf16*)(b1buf + lc * LINEB + half * 32);
      a1[j] = wmma_bf16(a, cat8(*(const bf16x8*)p1, *(const bf16x8*)(p1 + 8)), a1[j]);
      const bf16* p3 = (const bf16*)(b3buf + lc * LINEB + half * 32);
      a3[j] = wmma_bf16(a, cat8(*(const bf16x8*)p3, *(const bf16x8*)(p3 + 8)), a3[j]);
    }
    __syncthreads();                     // everyone done reading buffer (it&1)
    if (wave == 0 && it + 2 < NIT) {
      const unsigned dst = lds0 + (unsigned)((it & 1) * 2) * TILEB;
      tdm_load_tile_2d(dst,         B1 + k0 + 64, CH, CF, 32, 256, CH);
      tdm_load_tile_2d(dst + TILEB, B3 + k0 + 64, CH, CF, 32, 256, CH);
    }
  }
#else
  for (int k0 = 0; k0 < CH; k0 += 32) {
    bf16x16 a = cat8(*(const bf16x8*)(aRow + k0 + half * 16),
                     *(const bf16x8*)(aRow + k0 + half * 16 + 8));
#pragma unroll
    for (int j = 0; j < 4; ++j) {
      const bf16* bp1 = B1 + (size_t)(col0loc + j * 16 + ln) * CH + k0 + half * 16;
      a1[j] = wmma_bf16(a, cat8(*(const bf16x8*)bp1, *(const bf16x8*)(bp1 + 8)), a1[j]);
      const bf16* bp3 = B3 + (size_t)(col0loc + j * 16 + ln) * CH + k0 + half * 16;
      a3[j] = wmma_bf16(a, cat8(*(const bf16x8*)bp3, *(const bf16x8*)(bp3 + 8)), a3[j]);
    }
  }
#endif

#pragma unroll
  for (int j = 0; j < 4; ++j)
#pragma unroll
    for (int r = 0; r < 8; ++r) {
      float x1 = a1[j][r], x3 = a3[j][r];
      float g = x1 / (1.f + __expf(-x1));  // SiLU
      Act[(size_t)(row0 + r + 8 * half) * CF + colB + col0loc + j * 16 + ln] =
          (bf16)(g * x3);
    }
}

__global__ __launch_bounds__(256) void moe_gemm2_kernel(
    const bf16* __restrict__ Act, const bf16* __restrict__ w2T,
    const int* __restrict__ tile_expert, float* __restrict__ Yg) {
  const int e = tile_expert[blockIdx.y];
  if (e < 0) return;
  const int wave = threadIdx.x >> 5, lane = threadIdx.x & 31;
  const int half = lane >> 4, ln = lane & 15;
  const int row0 = blockIdx.y * 32 + (wave >> 2) * 16;
  const int colB = blockIdx.x * 256;
  const int col0loc = (wave & 3) * 64;

  const bf16* B2 = w2T + (size_t)e * CH * CF + (size_t)colB * CF;
  f32x8 acc[4];
#pragma unroll
  for (int j = 0; j < 4; ++j) acc[j] = zero8();
  const bf16* aRow = Act + (size_t)(row0 + ln) * CF;

#if HAVE_TDM
  constexpr int LINEB = 80;
  constexpr int TILEB = 256 * LINEB;
  __shared__ __attribute__((aligned(128))) char ldsB[2][TILEB];
  const int NIT = CF / 32;
  const unsigned lds0 = (unsigned)(uintptr_t)&ldsB[0][0];
  if (wave == 0) {
    tdm_load_tile_2d(lds0,         B2,      CF, CH, 32, 256, CF);
    tdm_load_tile_2d(lds0 + TILEB, B2 + 32, CF, CH, 32, 256, CF);
  }
  for (int it = 0; it < NIT; ++it) {
    const int k0 = it * 32;
    if (wave == 0) {
      if (it < NIT - 1) __builtin_amdgcn_s_wait_tensorcnt(1);
      else              __builtin_amdgcn_s_wait_tensorcnt(0);
    }
    __syncthreads();
    bf16x16 a = cat8(*(const bf16x8*)(aRow + k0 + half * 16),
                     *(const bf16x8*)(aRow + k0 + half * 16 + 8));
    const char* bbuf = &ldsB[it & 1][0];
#pragma unroll
    for (int j = 0; j < 4; ++j) {
      const int lc = col0loc + j * 16 + ln;
      const bf16* p = (const bf16*)(bbuf + lc * LINEB + half * 32);
      acc[j] = wmma_bf16(a, cat8(*(const bf16x8*)p, *(const bf16x8*)(p + 8)), acc[j]);
    }
    __syncthreads();
    if (wave == 0 && it + 2 < NIT)
      tdm_load_tile_2d(lds0 + (unsigned)(it & 1) * TILEB, B2 + k0 + 64,
                       CF, CH, 32, 256, CF);
  }
#else
  for (int k0 = 0; k0 < CF; k0 += 32) {
    bf16x16 a = cat8(*(const bf16x8*)(aRow + k0 + half * 16),
                     *(const bf16x8*)(aRow + k0 + half * 16 + 8));
#pragma unroll
    for (int j = 0; j < 4; ++j) {
      const bf16* bp = B2 + (size_t)(col0loc + j * 16 + ln) * CF + k0 + half * 16;
      acc[j] = wmma_bf16(a, cat8(*(const bf16x8*)bp, *(const bf16x8*)(bp + 8)), acc[j]);
    }
  }
#endif

#pragma unroll
  for (int j = 0; j < 4; ++j)
#pragma unroll
    for (int r = 0; r < 8; ++r)
      Yg[(size_t)(row0 + r + 8 * half) * CH + colB + col0loc + j * 16 + ln] = acc[j][r];
}

__global__ __launch_bounds__(256) void combine_kernel(
    const float* __restrict__ hidden1, const float* __restrict__ Yg,
    const int* __restrict__ tok2row, const float* __restrict__ topw,
    float* __restrict__ out) {
  const int t = blockIdx.x;
  int r0 = tok2row[t * 2], r1 = tok2row[t * 2 + 1];
  float w0 = topw[t * 2], w1 = topw[t * 2 + 1];
  for (int j = threadIdx.x; j < CH; j += 256)
    out[(size_t)t * CH + j] = hidden1[(size_t)t * CH + j] +
                              w0 * Yg[(size_t)r0 * CH + j] +
                              w1 * Yg[(size_t)r1 * CH + j];
}

__global__ __launch_bounds__(256) void copy_pos_kernel(const int* __restrict__ pos,
                                                       int* __restrict__ out) {
  int i = blockIdx.x * blockDim.x + threadIdx.x;
  if (i < CT) out[i] = pos[i];
}

// ------------------------------------------------------------------ launch ----
extern "C" void kernel_launch(void* const* d_in, const int* in_sizes, int n_in,
                              void* d_out, int out_size, void* d_ws, size_t ws_size,
                              hipStream_t stream) {
  (void)in_sizes; (void)n_in; (void)out_size; (void)ws_size;
  const float* hs  = (const float*)d_in[0];
  const int*   pos = (const int*)d_in[1];
  const float* ln1 = (const float*)d_in[3];
  const float* ln2 = (const float*)d_in[4];
  const float* wq  = (const float*)d_in[5];
  const float* wk  = (const float*)d_in[6];
  const float* wv  = (const float*)d_in[7];
  const float* wo  = (const float*)d_in[8];
  const float* rw  = (const float*)d_in[9];
  const float* w1  = (const float*)d_in[10];
  const float* w2  = (const float*)d_in[11];
  const float* w3  = (const float*)d_in[12];

  char* wp = (char*)d_ws;
  auto take = [&](size_t bytes) -> char* {
    char* p = wp;
    wp += (bytes + 255) & ~(size_t)255;
    return p;
  };
  bf16*  wqT  = (bf16*)take((size_t)CH * CH * 2);
  bf16*  wkT  = (bf16*)take((size_t)(CNKV * CHD) * CH * 2);
  bf16*  wvT  = (bf16*)take((size_t)(CNKV * CHD) * CH * 2);
  bf16*  woT  = (bf16*)take((size_t)CH * CH * 2);
  bf16*  w1T  = (bf16*)take((size_t)CE * CF * CH * 2);
  bf16*  w3T  = (bf16*)take((size_t)CE * CF * CH * 2);
  bf16*  w2T  = (bf16*)take((size_t)CE * CH * CF * 2);
  bf16*  xn   = (bf16*)take((size_t)CT * CH * 2);
  float* qf   = (float*)take((size_t)CT * CNH * CHD * 4);
  float* kf   = (float*)take((size_t)CT * CNKV * CHD * 4);
  float* vf   = (float*)take((size_t)CT * CNKV * CHD * 4);
  bf16*  qbb  = (bf16*)take((size_t)CT * CNH * CHD * 2);
  bf16*  kbb  = (bf16*)take((size_t)CT * CNKV * CHD * 2);
  bf16*  vbT  = (bf16*)take((size_t)CB * CNKV * CHD * CS * 2);
  bf16*  attnb= (bf16*)take((size_t)CT * CNH * CHD * 2);
  float* hidden1 = (float*)take((size_t)CT * CH * 4);
  bf16*  x2b  = (bf16*)take((size_t)CT * CH * 2);
  int*   topi = (int*)take((size_t)CT * 2 * 4);
  float* topw = (float*)take((size_t)CT * 2 * 4);
  int*   counts = (int*)take(CE * 4);
  int*   base   = (int*)take(CE * 4);
  int*   fill   = (int*)take(CE * 4);
  int*   tile_expert = (int*)take(MAXBT * 4);
  int*   tok2row = (int*)take((size_t)CT * 2 * 4);
  bf16*  Xg  = (bf16*)take((size_t)RMAX * CH * 2);
  bf16*  Act = (bf16*)take((size_t)RMAX * CF * 2);
  float* Yg  = (float*)take((size_t)RMAX * CH * 4);

  float* out_hidden = (float*)d_out;
  int*   out_pos    = (int*)((float*)d_out + (size_t)CT * CH);
  float* out_gate   = (float*)d_out + (size_t)CT * CH + CT;

  // --- weight conversion (fp32 -> bf16, transposed K-contiguous) ---
  {
    size_t t1 = (size_t)CH * CH;
    transpose_convert_kernel<<<(t1 + 255) / 256, 256, 0, stream>>>(wq, wqT, 1, CH, CH);
    size_t t2 = (size_t)CH * (CNKV * CHD);
    transpose_convert_kernel<<<(t2 + 255) / 256, 256, 0, stream>>>(wk, wkT, 1, CH, CNKV * CHD);
    transpose_convert_kernel<<<(t2 + 255) / 256, 256, 0, stream>>>(wv, wvT, 1, CH, CNKV * CHD);
    transpose_convert_kernel<<<(t1 + 255) / 256, 256, 0, stream>>>(wo, woT, 1, CNH * CHD, CH);
    size_t t3 = (size_t)CE * CH * CF;
    transpose_convert_kernel<<<(t3 + 255) / 256, 256, 0, stream>>>(w1, w1T, CE, CH, CF);
    transpose_convert_kernel<<<(t3 + 255) / 256, 256, 0, stream>>>(w3, w3T, CE, CH, CF);
    transpose_convert_kernel<<<(t3 + 255) / 256, 256, 0, stream>>>(w2, w2T, CE, CF, CH);
  }

  // --- attention block ---
  rmsnorm_kernel<<<CT, 256, 0, stream>>>(hs, ln1, xn);
  dim3 gq(CH / 256, CT / 32);
  gemm_bf16_kernel<0><<<gq, 256, 0, stream>>>(xn, wqT, qf, nullptr, CT, CH, CH);
  dim3 gkv((CNKV * CHD) / 256, CT / 32);
  gemm_bf16_kernel<0><<<gkv, 256, 0, stream>>>(xn, wkT, kf, nullptr, CT, CNKV * CHD, CH);
  gemm_bf16_kernel<0><<<gkv, 256, 0, stream>>>(xn, wvT, vf, nullptr, CT, CNKV * CHD, CH);
  {
    int tq = CT * CNH * (CHD / 2);
    rope_kernel<<<(tq + 255) / 256, 256, 0, stream>>>(qf, pos, qbb, CNH, 0.125f);
    int tk = CT * CNKV * (CHD / 2);
    rope_kernel<<<(tk + 255) / 256, 256, 0, stream>>>(kf, pos, kbb, CNKV, 1.0f);
    int tv = CT * CNKV * CHD;
    vtrans_kernel<<<(tv + 255) / 256, 256, 0, stream>>>(vf, vbT);
  }
  dim3 ga(CS / 16, CNH, CB);
  flash_attn_kernel<<<ga, 32, 0, stream>>>(qbb, kbb, vbT, attnb);
  gemm_bf16_kernel<1><<<gq, 256, 0, stream>>>(attnb, woT, hidden1, hs, CT, CH, CH);

  // --- MoE block ---
  rmsnorm_kernel<<<CT, 256, 0, stream>>>(hidden1, ln2, x2b);
  zero_u32_kernel<<<1, 32, 0, stream>>>((unsigned*)counts, CE);
  zero_u32_kernel<<<1, 32, 0, stream>>>((unsigned*)fill, CE);
  router_kernel<<<CT, 256, 0, stream>>>(x2b, rw, out_gate, topi, topw, counts);
  offsets_kernel<<<1, 32, 0, stream>>>(counts, base, tile_expert);
  scatter_kernel<<<(CT * 2 + 255) / 256, 256, 0, stream>>>(topi, base, fill, tok2row);
  {
    size_t nw = (size_t)RMAX * CH * 2 / 4;  // zero Xg (pad rows must be 0)
    zero_u32_kernel<<<(nw + 255) / 256, 256, 0, stream>>>((unsigned*)Xg, nw);
  }
  gather_kernel<<<CT * 2, 256, 0, stream>>>(x2b, tok2row, Xg);
  dim3 g13(CF / 256, MAXBT);
  moe_gemm13_kernel<<<g13, 256, 0, stream>>>(Xg, w1T, w3T, tile_expert, Act);
  dim3 g2(CH / 256, MAXBT);
  moe_gemm2_kernel<<<g2, 256, 0, stream>>>(Act, w2T, tile_expert, Yg);
  combine_kernel<<<CT, 256, 0, stream>>>(hidden1, Yg, tok2row, topw, out_hidden);
  copy_pos_kernel<<<(CT + 255) / 256, 256, 0, stream>>>(pos, out_pos);
}
